// ROEN_32066225832591
// MI455X (gfx1250) — compile-verified, matrix operand order checked by reading, not run on hardware
//
#include <hip/hip_runtime.h>

// ---------------------------------------------------------------------------
// MI455X (gfx1250, wave32) temporal-GNN. Dense layers on V_WMMA_F32_16X16X32_F16
// (16-row tiles/wave, f32 accum, LDS-staged transposed f16 weights, hoisted
// biases). Inception staging uses GLOBAL_LOAD_ASYNC_TO_LDS (ASYNCcnt).
// Scatter-softmax attention via sortable-uint atomicMax + float atomicAdd.
// ---------------------------------------------------------------------------

#define T_  8
#define N_  20000
#define E_  100000
#define H_  4      // attention heads
#define CD_ 32     // head dim

typedef __attribute__((ext_vector_type(16))) _Float16 v16h;
typedef __attribute__((ext_vector_type(8)))  _Float16 v8h;
typedef __attribute__((ext_vector_type(8)))  float    v8f;

// ------------------------- WMMA fragment helpers ---------------------------
// 16-bit A 16x32 layout (ISA 7.12.2): lane = half*16 + m; within a lane,
// f16 elements 0..7 hold K = half*8 .. half*8+7 (contiguous), elements 8..15
// hold K = 16+half*8 .. 16+half*8+7.  => A frag = 4x b128 (f32) / 2x b128 (f16).

template <int KV>
__device__ __forceinline__ v16h frag_a_ptr(const float* __restrict__ rp) {
  const int half = (threadIdx.x >> 4) & 1;
  v16h a;
  float4 u0 = *(const float4*)(rp + half * 8);
  float4 u1 = *(const float4*)(rp + half * 8 + 4);
  a[0] = (_Float16)u0.x; a[1] = (_Float16)u0.y;
  a[2] = (_Float16)u0.z; a[3] = (_Float16)u0.w;
  a[4] = (_Float16)u1.x; a[5] = (_Float16)u1.y;
  a[6] = (_Float16)u1.z; a[7] = (_Float16)u1.w;
  if (KV >= 32) {
    float4 u2 = *(const float4*)(rp + 16 + half * 8);
    float4 u3 = *(const float4*)(rp + 16 + half * 8 + 4);
    a[8]  = (_Float16)u2.x; a[9]  = (_Float16)u2.y;
    a[10] = (_Float16)u2.z; a[11] = (_Float16)u2.w;
    a[12] = (_Float16)u3.x; a[13] = (_Float16)u3.y;
    a[14] = (_Float16)u3.z; a[15] = (_Float16)u3.w;
  } else {
#pragma unroll
    for (int i = 8; i < 16; ++i) a[i] = (_Float16)0.f;
  }
  return a;
}

template <int KV>
__device__ __forceinline__ v16h frag_a_rows(const float* __restrict__ src, long ld,
                                            long row0) {
  return frag_a_ptr<KV>(src + (row0 + (threadIdx.x & 15)) * ld);
}

// Fragment from an f16 LDS tile (row-major, leading dim ldl, K offset k0).
// Serves as A fragment (row per lane) AND as B fragment when the tile holds
// the TRANSPOSED weight W^T[N][K] (lane n reads contiguous K of column n).
__device__ __forceinline__ v16h frag_lds(const _Float16* __restrict__ st, int ldl,
                                         int k0) {
  const int lane = threadIdx.x & 31;
  const int half = lane >> 4, m = lane & 15;
  const _Float16* p = st + m * ldl + k0 + half * 8;
  v8h h0 = *(const v8h*)p;
  v8h h1 = *(const v8h*)(p + 16);
  v16h a;
#pragma unroll
  for (int i = 0; i < 8; ++i) { a[i] = h0[i]; a[8 + i] = h1[i]; }
  return a;
}

// C/D init by broadcasting a pre-loaded per-lane bias value.
__device__ __forceinline__ v8f frag_c_val(float bv) {
  v8f c;
#pragma unroll
  for (int r = 0; r < 8; ++r) c[r] = bv;
  return c;
}

__device__ __forceinline__ v8f wmma32(v16h a, v16h b, v8f c) {
  return __builtin_amdgcn_wmma_f32_16x16x32_f16(false, a, false, b, (short)0, c, false, false);
}

template <int NV>
__device__ __forceinline__ void store_c_global(float* __restrict__ dst, long ld,
                                               long row0, int n0, v8f c, bool relu) {
  int lane = threadIdx.x & 31, half = lane >> 4, n = lane & 15;
  if (n >= NV) return;
#pragma unroll
  for (int r = 0; r < 8; ++r) {
    float v = c[r];
    if (relu) v = fmaxf(v, 0.f);
    dst[(row0 + r + 8 * half) * ld + n0 + n] = v;
  }
}

__device__ __forceinline__ void store_c_lds(_Float16* __restrict__ st, int ldl, int n0,
                                            v8f c, bool relu) {
  int lane = threadIdx.x & 31, half = lane >> 4, n = lane & 15;
#pragma unroll
  for (int r = 0; r < 8; ++r) {
    float v = c[r];
    if (relu) v = fmaxf(v, 0.f);
    st[(r + 8 * half) * ldl + n0 + n] = (_Float16)v;
  }
}

// Cooperatively stage W[K,Nn] (row-major f32) transposed into WT[NP][KP] f16,
// zero-padding n>=Nn / k>=K.  Call before __syncthreads().
__device__ __forceinline__ void stage_wt(_Float16* __restrict__ wt,
                                         const float* __restrict__ W,
                                         int K, int Nn, int NP, int KP) {
  for (int idx = threadIdx.x; idx < NP * KP; idx += blockDim.x) {
    int n = idx / KP, k = idx % KP;
    wt[idx] = (_Float16)((k < K && n < Nn) ? W[(long)k * Nn + n] : 0.f);
  }
}

// sortable-uint encoding for float atomicMax segment-max
__device__ __forceinline__ unsigned enc_f(float f) {
  unsigned u = __float_as_uint(f);
  return (u & 0x80000000u) ? ~u : (u | 0x80000000u);
}
__device__ __forceinline__ float dec_f(unsigned u) {
  return __uint_as_float((u & 0x80000000u) ? (u ^ 0x80000000u) : ~u);
}

// ---- CDNA5 async global->LDS (per-lane gather, tracked by ASYNCcnt) -------
// LDS dest address = low 32 bits of the generic pointer (ISA 10.2: for LDS-
// aperture addresses).
__device__ __forceinline__ void async_g2l_b32(unsigned lds_off,
                                              const void* __restrict__ gptr) {
  unsigned long long ga = (unsigned long long)gptr;
  asm volatile("global_load_async_to_lds_b32 %0, %1, off"
               :: "v"(lds_off), "v"(ga) : "memory");
}
__device__ __forceinline__ void wait_asynccnt0() {
  asm volatile("s_wait_asynccnt 0x0" ::: "memory");
}

// ---------------------------------------------------------------------------
// Fused 2-layer MLP (+ optional 3rd linear), ReLU after every layer.
// FIN in {16,32}; hidden widths 32. 16 rows/wave, 8 waves/block.
// ---------------------------------------------------------------------------
template <int FIN, bool HAS3>
__global__ void k_mlp2(const float* __restrict__ X,
                       const float* __restrict__ W1, const float* __restrict__ B1,
                       const float* __restrict__ W2, const float* __restrict__ B2,
                       float* __restrict__ Y1,
                       const float* __restrict__ W3, const float* __restrict__ B3,
                       float* __restrict__ Y2) {
  __shared__ __align__(16) _Float16 stage[8][16][32];
  __shared__ __align__(16) _Float16 w1t[32][32];
  __shared__ __align__(16) _Float16 w2t[32][32];
  __shared__ __align__(16) _Float16 w3t[HAS3 ? 32 : 1][HAS3 ? 32 : 1];
  stage_wt(&w1t[0][0], W1, FIN, 32, 32, 32);
  stage_wt(&w2t[0][0], W2, 32, 32, 32, 32);
  if (HAS3) stage_wt(&w3t[0][0], W3, 32, 32, 32, 32);

  // hoist biases (per-lane column value), overlapping with staging
  int n = threadIdx.x & 15;
  float b1a = B1[n], b1b = B1[16 + n];
  float b2a = B2[n], b2b = B2[16 + n];
  float b3a = 0.f, b3b = 0.f;
  if (HAS3) { b3a = B3[n]; b3b = B3[16 + n]; }
  __syncthreads();

  int wv = threadIdx.x >> 5;
  long row0 = ((long)blockIdx.x * 8 + wv) * 16;

  // layer 1: FIN -> 32
  v16h a = frag_a_rows<FIN>(X, FIN, row0);
  v8f c0 = wmma32(a, frag_lds(&w1t[0][0], 32, 0), frag_c_val(b1a));
  v8f c1 = wmma32(a, frag_lds(&w1t[16][0], 32, 0), frag_c_val(b1b));
  store_c_lds(&stage[wv][0][0], 32, 0, c0, true);
  store_c_lds(&stage[wv][0][0], 32, 16, c1, true);
  __syncthreads();

  // layer 2: 32 -> 32
  v16h a2 = frag_lds(&stage[wv][0][0], 32, 0);
  v8f d0 = wmma32(a2, frag_lds(&w2t[0][0], 32, 0), frag_c_val(b2a));
  v8f d1 = wmma32(a2, frag_lds(&w2t[16][0], 32, 0), frag_c_val(b2b));
  store_c_global<16>(Y1, 32, row0, 0, d0, true);
  store_c_global<16>(Y1, 32, row0, 16, d1, true);

  if (HAS3) {  // layer 3: 32 -> 32 (edge-feature branch ef)
    __syncthreads();
    store_c_lds(&stage[wv][0][0], 32, 0, d0, true);
    store_c_lds(&stage[wv][0][0], 32, 16, d1, true);
    __syncthreads();
    v16h a3 = frag_lds(&stage[wv][0][0], 32, 0);
    v8f e0 = wmma32(a3, frag_lds(&w3t[0][0], 32, 0), frag_c_val(b3a));
    v8f e1 = wmma32(a3, frag_lds(&w3t[16][0], 32, 0), frag_c_val(b3b));
    store_c_global<16>(Y2, 32, row0, 0, e0, true);
    store_c_global<16>(Y2, 32, row0, 16, e1, true);
  }
}

// ---------------------------------------------------------------------------
// Q/K/V (32->128 each) + skip (32->32) projections, all T. 26 WMMA/wave.
// qkv layout: [T*N, 384] = [q | k | v].
// ---------------------------------------------------------------------------
__global__ void k_proj(const float* __restrict__ Hin,
                       const float* __restrict__ wq, const float* __restrict__ bq,
                       const float* __restrict__ wk, const float* __restrict__ bk,
                       const float* __restrict__ wv, const float* __restrict__ bv,
                       const float* __restrict__ wsk, const float* __restrict__ bsk,
                       float* __restrict__ QKV, float* __restrict__ SKIP) {
  __shared__ __align__(16) _Float16 wqt[128][32];
  __shared__ __align__(16) _Float16 wkt[128][32];
  __shared__ __align__(16) _Float16 wvt[128][32];
  __shared__ __align__(16) _Float16 wst[32][32];
  stage_wt(&wqt[0][0], wq, 32, 128, 128, 32);
  stage_wt(&wkt[0][0], wk, 32, 128, 128, 32);
  stage_wt(&wvt[0][0], wv, 32, 128, 128, 32);
  stage_wt(&wst[0][0], wsk, 32, 32, 32, 32);

  // hoist all bias values before the WMMA loop
  int n = threadIdx.x & 15;
  float bqv[8], bkv[8], bvv[8], bsv[2];
#pragma unroll
  for (int i = 0; i < 8; ++i) {
    bqv[i] = bq[i * 16 + n];
    bkv[i] = bk[i * 16 + n];
    bvv[i] = bv[i * 16 + n];
  }
  bsv[0] = bsk[n]; bsv[1] = bsk[16 + n];
  __syncthreads();

  int wvx = threadIdx.x >> 5;
  long row0 = ((long)blockIdx.x * 8 + wvx) * 16;
  v16h a = frag_a_rows<32>(Hin, 32, row0);
#pragma unroll
  for (int n0 = 0; n0 < 8; ++n0) {
    v8f cq = wmma32(a, frag_lds(&wqt[n0 * 16][0], 32, 0), frag_c_val(bqv[n0]));
    store_c_global<16>(QKV, 384, row0, n0 * 16, cq, false);
    v8f ck = wmma32(a, frag_lds(&wkt[n0 * 16][0], 32, 0), frag_c_val(bkv[n0]));
    store_c_global<16>(QKV, 384, row0, 128 + n0 * 16, ck, false);
    v8f cv = wmma32(a, frag_lds(&wvt[n0 * 16][0], 32, 0), frag_c_val(bvv[n0]));
    store_c_global<16>(QKV, 384, row0, 256 + n0 * 16, cv, false);
  }
#pragma unroll
  for (int n0 = 0; n0 < 2; ++n0) {
    v8f cs = wmma32(a, frag_lds(&wst[n0 * 16][0], 32, 0), frag_c_val(bsv[n0]));
    store_c_global<16>(SKIP, 32, row0, n0 * 16, cs, false);
  }
}

// ---- attention scratch init: acc=0, s=0, m=enc(-inf)~0 ----
__global__ void k_init(float* __restrict__ ACC, float* __restrict__ S,
                       unsigned* __restrict__ M) {
  if (__builtin_amdgcn_cluster_id_x() > 15) return;  // never taken; NOP w/o cluster
  long idx = (long)blockIdx.x * blockDim.x + threadIdx.x;  // covers T*N*128
  ACC[idx] = 0.f;
  if (idx < (long)T_ * N_ * H_) { S[idx] = 0.f; M[idx] = 0u; }
}

// ---- pass 1: per-(t,e,h) logits + segment max (atomicMax on encoded) ----
__global__ void k_logits(const int* __restrict__ EI, const float* __restrict__ QKV,
                         const float* __restrict__ EA, const float* __restrict__ WE,
                         float* __restrict__ LOG, unsigned* __restrict__ M) {
  long idx = (long)blockIdx.x * blockDim.x + threadIdx.x;  // (t*H + h)*E + e
  int e = (int)(idx % E_);
  int h = (int)((idx / E_) % H_);
  int t = (int)(idx / ((long)E_ * H_));
  const int* ei = EI + (long)t * 2 * E_;
  int sN = ei[e], dN = ei[E_ + e];
  const float* q  = QKV + ((long)t * N_ + dN) * 384 + h * CD_;
  const float* kk = QKV + ((long)t * N_ + sN) * 384 + 128 + h * CD_;
  const float* ez = EA + ((long)t * E_ + e) * 32;
  float qr[32], kr[32], er[32];
#pragma unroll
  for (int i = 0; i < 8; ++i) {
    *(float4*)(qr + 4 * i) = *(const float4*)(q + 4 * i);
    *(float4*)(kr + 4 * i) = *(const float4*)(kk + 4 * i);
    *(float4*)(er + 4 * i) = *(const float4*)(ez + 4 * i);
  }
  float dot = 0.f;
#pragma unroll 4
  for (int c = 0; c < CD_; ++c) {
    const float* wc = WE + h * CD_ + c;  // WE[32,128]; wave-uniform -> SALU
    float ep = 0.f;
#pragma unroll
    for (int j = 0; j < 32; ++j) ep += er[j] * wc[(long)j * 128];
    dot += qr[c] * (kr[c] + ep);
  }
  float lg = dot * 0.17677669529663687f;  // 1/sqrt(32)
  LOG[((long)t * E_ + e) * H_ + h] = lg;
  atomicMax(&M[((long)t * N_ + dN) * H_ + h], enc_f(lg));
}

// ---- pass 2: exp + un-normalized weighted value scatter ----
__global__ void k_scatter(const int* __restrict__ EI, const float* __restrict__ QKV,
                          const float* __restrict__ EA, const float* __restrict__ WE,
                          const float* __restrict__ LOG, const unsigned* __restrict__ M,
                          float* __restrict__ S, float* __restrict__ ACC) {
  long idx = (long)blockIdx.x * blockDim.x + threadIdx.x;
  int e = (int)(idx % E_);
  int h = (int)((idx / E_) % H_);
  int t = (int)(idx / ((long)E_ * H_));
  const int* ei = EI + (long)t * 2 * E_;
  int sN = ei[e], dN = ei[E_ + e];
  float lg = LOG[((long)t * E_ + e) * H_ + h];
  float mm = dec_f(M[((long)t * N_ + dN) * H_ + h]);
  float ex = __expf(lg - mm);
  atomicAdd(&S[((long)t * N_ + dN) * H_ + h], ex);
  const float* vv = QKV + ((long)t * N_ + sN) * 384 + 256 + h * CD_;
  const float* ez = EA + ((long)t * E_ + e) * 32;
  float vr[32], er[32];
#pragma unroll
  for (int i = 0; i < 8; ++i) {
    *(float4*)(vr + 4 * i) = *(const float4*)(vv + 4 * i);
    *(float4*)(er + 4 * i) = *(const float4*)(ez + 4 * i);
  }
  float* accp = ACC + (((long)t * N_ + dN) * H_ + h) * CD_;
#pragma unroll 4
  for (int c = 0; c < CD_; ++c) {
    const float* wc = WE + h * CD_ + c;
    float ep = 0.f;
#pragma unroll
    for (int j = 0; j < 32; ++j) ep += er[j] * wc[(long)j * 128];
    atomicAdd(&accp[c], ex * (vr[c] + ep));
  }
}

// ---- pass 3: normalize, head mean, + skip, optional ReLU ----
__global__ void k_finalize(const float* __restrict__ ACC, const float* __restrict__ S,
                           const float* __restrict__ SKIP, int relu,
                           float* __restrict__ OUT) {
  long idx = (long)blockIdx.x * blockDim.x + threadIdx.x;  // T*N*32
  int c = (int)(idx & 31);
  long tn = idx >> 5;
  float o = 0.f;
#pragma unroll
  for (int h = 0; h < H_; ++h) {
    float sv = S[tn * H_ + h];
    if (sv > 0.f) o += ACC[(tn * H_ + h) * CD_ + c] / sv;
  }
  o = o * (1.f / H_) + SKIP[tn * 32 + c];
  OUT[tn * 32 + c] = relu ? fmaxf(o, 0.f) : o;
}

// ---------------------------------------------------------------------------
// Temporal inception: causal dilated conv1d over T=8, kernels {1,2,3,5},
// dilation 2, 8 out-channels each, identity residual, ReLU. 8 nodes/block.
// LDS staging via CDNA5 GLOBAL_LOAD_ASYNC_TO_LDS_B32 (ASYNCcnt).
// ---------------------------------------------------------------------------
__global__ void k_incep(const float* __restrict__ H2,
                        const float* __restrict__ tw1, const float* __restrict__ tb1,
                        const float* __restrict__ tw2, const float* __restrict__ tb2,
                        const float* __restrict__ tw3, const float* __restrict__ tb3,
                        const float* __restrict__ tw5, const float* __restrict__ tb5,
                        float* __restrict__ NOUT) {
  __shared__ float xs[8][32][8];  // [node][cin][t]
  int ln = threadIdx.x >> 5, c = threadIdx.x & 31;
  long n = (long)blockIdx.x * 8 + ln;
  // async per-lane gather into LDS: xs[ln][c][t] = H2[(t*N + n)*32 + c]
  unsigned lds0 = (unsigned)(unsigned long long)(const void*)&xs[ln][c][0];
#pragma unroll
  for (int t = 0; t < T_; ++t)
    async_g2l_b32(lds0 + 4u * t, (const void*)(H2 + ((long)t * N_ + n) * 32 + c));
  wait_asynccnt0();
  __syncthreads();
  int g = c >> 3, co = c & 7;
  const float* tw; const float* tb; int kk;
  if (g == 0)      { tw = tw1; tb = tb1; kk = 1; }
  else if (g == 1) { tw = tw2; tb = tb2; kk = 2; }
  else if (g == 2) { tw = tw3; tb = tb3; kk = 3; }
  else             { tw = tw5; tb = tb5; kk = 5; }
  float bias = tb[co];
#pragma unroll
  for (int t = 0; t < T_; ++t) {
    float acc = bias;
    for (int j = 0; j < kk; ++j) {
      int tp = t - 2 * (kk - 1 - j);  // dilation 2, causal left pad
      if (tp < 0) continue;
      const float* wj = tw + (long)co * 32 * kk + j;  // tc_w[co][ci][j]
#pragma unroll
      for (int ci = 0; ci < 32; ++ci) acc += wj[(long)ci * kk] * xs[ln][ci][tp];
    }
    float r = acc + xs[ln][c][t];  // identity residual
    NOUT[((long)t * N_ + n) * 32 + c] = fmaxf(r, 0.f);
  }
}

// ---------------------------------------------------------------------------
// Classifier: per edge concat [nout[src] | nout[dst] | ef] (96) -> 128 -> 8.
// 16 edges/wave: 3 K-chunks x 8 N-tiles + 4 WMMA layer 2 = 28 WMMA/wave.
// ---------------------------------------------------------------------------
__global__ void k_cls(const float* __restrict__ NOUT, const float* __restrict__ EF,
                      const int* __restrict__ EI,
                      const float* __restrict__ W1, const float* __restrict__ B1,
                      const float* __restrict__ W2, const float* __restrict__ B2,
                      float* __restrict__ OUT) {
  __shared__ __align__(16) _Float16 stage[8][16][128];  // 32 KB
  __shared__ __align__(16) _Float16 w1t[128][96];       // 24 KB (W1^T)
  __shared__ __align__(16) _Float16 w2t[16][128];       // 4 KB  (W2^T, rows 8..15 zero)
  stage_wt(&w1t[0][0], W1, 96, 128, 128, 96);
  stage_wt(&w2t[0][0], W2, 128, 8, 16, 128);

  int nlane = threadIdx.x & 15;
  float b1v[8];
#pragma unroll
  for (int i = 0; i < 8; ++i) b1v[i] = B1[i * 16 + nlane];
  float b2v = (nlane < 8) ? B2[nlane] : 0.f;
  __syncthreads();

  int wv = threadIdx.x >> 5, lane = threadIdx.x & 31, m = lane & 15;
  long wid = (long)blockIdx.x * 8 + wv;  // E/16 = 6250 waves per t
  int t = (int)(wid / (E_ / 16));
  long e0 = (wid % (E_ / 16)) * 16;
  const int* ei = EI + (long)t * 2 * E_;
  int es = ei[e0 + m], ed = ei[E_ + e0 + m];
  const float* r_src = NOUT + ((long)t * N_ + es) * 32;
  const float* r_dst = NOUT + ((long)t * N_ + ed) * 32;
  const float* r_ef  = EF + ((long)t * E_ + e0 + m) * 32;
  __builtin_prefetch(r_src, 0, 1);
  __builtin_prefetch(r_dst, 0, 1);
  v16h a0 = frag_a_ptr<32>(r_src);
  v16h a1 = frag_a_ptr<32>(r_dst);
  v16h a2 = frag_a_ptr<32>(r_ef);
#pragma unroll
  for (int n0 = 0; n0 < 8; ++n0) {
    v8f cc = frag_c_val(b1v[n0]);
    cc = wmma32(a0, frag_lds(&w1t[n0 * 16][0], 96, 0), cc);
    cc = wmma32(a1, frag_lds(&w1t[n0 * 16][0], 96, 32), cc);
    cc = wmma32(a2, frag_lds(&w1t[n0 * 16][0], 96, 64), cc);
    store_c_lds(&stage[wv][0][0], 128, n0 * 16, cc, true);
  }
  __syncthreads();
  v8f o = frag_c_val(b2v);
#pragma unroll
  for (int kc = 0; kc < 4; ++kc) {
    v16h aa = frag_lds(&stage[wv][0][0], 128, kc * 32);
    o = wmma32(aa, frag_lds(&w2t[0][0], 128, kc * 32), o);
  }
  store_c_global<8>(OUT, 8, (long)t * E_ + e0, 0, o, false);
  __builtin_amdgcn_s_wait_tensorcnt(0);  // CDNA5 split-counter wait (no-op here)
}

// ---------------------------------------------------------------------------
extern "C" void kernel_launch(void* const* d_in, const int* in_sizes, int n_in,
                              void* d_out, int out_size, void* d_ws, size_t ws_size,
                              hipStream_t stream) {
  (void)in_sizes; (void)n_in; (void)out_size; (void)ws_size;
  // setup_inputs() dict order, params flattened in insertion order:
  const float* x   = (const float*)d_in[0];   // [T,N,32]
  const float* eat = (const float*)d_in[1];   // [T,E,16]
  const int*   ei  = (const int*)d_in[2];     // [T,2,E]
  const float* mn_w1 = (const float*)d_in[3];
  const float* mn_b1 = (const float*)d_in[4];
  const float* mn_w2 = (const float*)d_in[5];
  const float* mn_b2 = (const float*)d_in[6];
  const float* me_w1 = (const float*)d_in[7];
  const float* me_b1 = (const float*)d_in[8];
  const float* me_w2 = (const float*)d_in[9];
  const float* me_b2 = (const float*)d_in[10];
  // g1: wq,bq,wk,bk,wv,bv,we,ws,bs = 11..19 ; g2 = 20..28
  const float* ef_w = (const float*)d_in[29];
  const float* ef_b = (const float*)d_in[30];
  const float* tw1 = (const float*)d_in[31]; const float* tb1 = (const float*)d_in[32];
  const float* tw2 = (const float*)d_in[33]; const float* tb2 = (const float*)d_in[34];
  const float* tw3 = (const float*)d_in[35]; const float* tb3 = (const float*)d_in[36];
  const float* tw5 = (const float*)d_in[37]; const float* tb5 = (const float*)d_in[38];
  const float* cl_w1 = (const float*)d_in[39]; const float* cl_b1 = (const float*)d_in[40];
  const float* cl_w2 = (const float*)d_in[41]; const float* cl_b2 = (const float*)d_in[42];

  const long TN = (long)T_ * N_;   // 160000
  const long TE = (long)T_ * E_;   // 800000
  float* ws = (float*)d_ws;
  long off = 0;
  float* h0   = ws + off; off += TN * 32;
  float* ea   = ws + off; off += TE * 32;
  float* ef   = ws + off; off += TE * 32;
  float* qkv  = ws + off; off += TN * 384;
  float* skip = ws + off; off += TN * 32;
  float* logi = ws + off; off += TE * H_;
  unsigned* menc = (unsigned*)(ws + off); off += TN * H_;
  float* svec = ws + off; off += TN * H_;
  float* acc  = ws + off; off += TN * H_ * CD_;
  float* h1   = ws + off; off += TN * 32;
  float* h2   = ws + off; off += TN * 32;
  float* nout = ws + off; off += TN * 32;

  dim3 B(256);
  // node MLP: [T*N,32] -> h0 ; 160000/(16*8)=1250 blocks
  k_mlp2<32, false><<<1250, B, 0, stream>>>(x, mn_w1, mn_b1, mn_w2, mn_b2, h0,
                                            nullptr, nullptr, nullptr);
  // edge MLP (+fused ef): [T*E,16] -> ea, ef ; 800000/128=6250 blocks
  k_mlp2<16, true><<<6250, B, 0, stream>>>(eat, me_w1, me_b1, me_w2, me_b2, ea,
                                           ef_w, ef_b, ef);

  for (int l = 0; l < 2; ++l) {
    const float* wq = (const float*)d_in[11 + l * 9];
    const float* bq = (const float*)d_in[12 + l * 9];
    const float* wk = (const float*)d_in[13 + l * 9];
    const float* bk = (const float*)d_in[14 + l * 9];
    const float* wv = (const float*)d_in[15 + l * 9];
    const float* bv = (const float*)d_in[16 + l * 9];
    const float* we = (const float*)d_in[17 + l * 9];
    const float* wsk = (const float*)d_in[18 + l * 9];
    const float* bsk = (const float*)d_in[19 + l * 9];
    const float* hin = (l == 0) ? h0 : h1;
    float* hout = (l == 0) ? h1 : h2;
    k_init<<<80000, B, 0, stream>>>(acc, svec, menc);  // T*N*128 threads
    k_proj<<<1250, B, 0, stream>>>(hin, wq, bq, wk, bk, wv, bv, wsk, bsk, qkv, skip);
    k_logits<<<12500, B, 0, stream>>>(ei, qkv, ea, we, logi, menc);  // T*H*E
    k_scatter<<<12500, B, 0, stream>>>(ei, qkv, ea, we, logi, menc, svec, acc);
    k_finalize<<<20000, B, 0, stream>>>(acc, svec, skip, (l == 0) ? 1 : 0, hout);
  }

  // temporal inception over h2 -> nout ; 20000/8 = 2500 blocks
  k_incep<<<2500, B, 0, stream>>>(h2, tw1, tb1, tw2, tb2, tw3, tb3, tw5, tb5, nout);
  // classifier -> out [T,E,8]
  k_cls<<<6250, B, 0, stream>>>(nout, ef, ei, cl_w1, cl_b1, cl_w2, cl_b2,
                                (float*)d_out);
}
